// LSTMModel_70368744178317
// MI455X (gfx1250) — matrix operand order repaired and checked by reading, not compiled, not run on hardware
//
#include <hip/hip_runtime.h>
#include <hip/hip_bf16.h>
#include <math.h>

// ---------------------------------------------------------------------------
// Sizes (fixed by the reference): T=8192, IN=9, H=1024, 4H=4096, OUT=1
// ---------------------------------------------------------------------------
#define H4 4096
#define HD 1024
#define L0WG 64          // persistent WGs running layer 0
#define L1WG 64          // persistent WGs running layer 1 (lagged by LAG steps)
#define NWGT (L0WG + L1WG)
#define UPW 16           // hidden units per WG  (64 * 16 == 1024)
#define ROWS 64          // gate rows per WG (4 gates * UPW)
#define WSTR 1032        // LDS row stride: 1032 % 64 == 8 -> conflict-free dots
#define LAG 16           // layer-1 pipeline lag == WMMA chunk (M dimension)

typedef _Float16 v8h  __attribute__((ext_vector_type(8)));
typedef _Float16 v16h __attribute__((ext_vector_type(16)));
typedef float    v8f  __attribute__((ext_vector_type(8)));

__device__ __forceinline__ float sigmoidf_(float x) {
    return 1.0f / (1.0f + __expf(-x));
}

// ---------------------------------------------------------------------------
// Zero the ping-pong h buffers and the grid-barrier counter.
// ---------------------------------------------------------------------------
__global__ void init_state(float* hbuf0, float* hbuf2, unsigned* counter) {
    int i = threadIdx.x;
    for (int k = i; k < 2 * HD; k += 256) { hbuf0[k] = 0.0f; hbuf2[k] = 0.0f; }
    if (i == 0) counter[0] = 0u;
}

// ---------------------------------------------------------------------------
// gx0[t][n] = x[t]·w_ih0[n] + b_ih0[n] + b_hh0[n]     (IN = 9, tiny K)
// ---------------------------------------------------------------------------
__global__ void gx0_kernel(const float* __restrict__ x,      // [T x 9]
                           const float* __restrict__ w,      // [4096 x 9]
                           const float* __restrict__ bi,
                           const float* __restrict__ bh,
                           float* __restrict__ gx, int T) {
    size_t i = (size_t)blockIdx.x * 256 + threadIdx.x;
    if (i >= (size_t)T * H4) return;
    int t = (int)(i >> 12);
    int n = (int)(i & (H4 - 1));
    float s = bi[n] + bh[n];
    const float* xr = x + (size_t)t * 9;
    const float* wr = w + (size_t)n * 9;
#pragma unroll
    for (int k = 0; k < 9; ++k) s = fmaf(xr[k], wr[k], s);
    gx[i] = s;
}

// ---------------------------------------------------------------------------
// fp32 -> f16 convert (w_ih1, so the chunked gx1 GEMM runs on the WMMA pipe)
// ---------------------------------------------------------------------------
__global__ void cvt_f16(const float* __restrict__ src, _Float16* __restrict__ dst,
                        size_t n) {
    size_t i = (size_t)blockIdx.x * 256 + threadIdx.x;
    if (i < n) dst[i] = (_Float16)src[i];
}

// ---------------------------------------------------------------------------
// Fused, software-pipelined 2-layer LSTM scan (persistent kernel).
//  - WGs [0,64):   layer 0, timestep t = s.      W_hh0 slice in LDS.
//  - WGs [64,128): layer 1, timestep t = s - 16. W_hh1 slice in LDS.
//    Every 16 steps, 4 waves compute the chunk's input gates
//    gx1[t..t+15][64 rows] = h1_chunk(f16) @ w_ih1_slice^T  (+ biases)
//    via v_wmma_f32_16x16x32_f16 into LDS.
//  One device-scope barrier per global step: T+LAG syncs total (vs 2T).
// ---------------------------------------------------------------------------
__global__ __launch_bounds__(256, 1)
void lstm_fused(const float* __restrict__ whh0,   // [4096 x 1024] fp32
                const float* __restrict__ gx0,    // [T x 4096] fp32 (w/ biases)
                const float* __restrict__ whh1,   // [4096 x 1024] fp32
                const _Float16* __restrict__ wih1h,// [4096 x 1024] f16
                const float* __restrict__ bih1,
                const float* __restrict__ bhh1,
                _Float16* __restrict__ h1h,       // [T x 1024] f16 (layer0 out)
                float* __restrict__ hbuf0,        // [2 x 1024] layer-0 ping-pong
                float* __restrict__ hbuf2,        // [2 x 1024] layer-1 ping-pong
                unsigned* __restrict__ counter,   // grid barrier epoch counter
                int T) {
    extern __shared__ float lds[];
    float* wl  = lds;                         // ROWS * WSTR   (W_hh slice)
    float* hl  = wl + ROWS * WSTR;            // HD            (staged h)
    float* gl  = hl + HD;                     // ROWS          (gate pre-acts)
    float* gxb = gl + ROWS;                   // LAG * ROWS    (layer-1 gx chunk)

    const int wg   = blockIdx.x;
    const bool l1  = (wg >= L0WG);
    const int wid  = l1 ? (wg - L0WG) : wg;   // 0..63 within role
    const int tid  = threadIdx.x;
    const int lane = tid & 31;
    const int wv   = tid >> 5;
    const int r    = tid >> 2;                // gate row 0..63 (4 lanes/row)
    const int sub  = tid & 3;

    // Stage this WG's W_hh slice into LDS (gate-major, unit-minor rows).
    const float* wsrc = l1 ? whh1 : whh0;
    for (int i = tid; i < ROWS * HD; i += 256) {
        int rr = i >> 10, k = i & (HD - 1);
        int R  = (rr >> 4) * HD + wid * UPW + (rr & 15);
        wl[rr * WSTR + k] = wsrc[(size_t)R * HD + k];
    }
    __syncthreads();

    const int Rglob = (r >> 4) * HD + wid * UPW + (r & 15);
    float* myh  = l1 ? hbuf2 : hbuf0;
    float c_reg = 0.0f;                       // cell state in a VGPR (tid<UPW)

    const int steps = T + LAG;
    for (int s = 0; s < steps; ++s) {
        const int t = l1 ? (s - LAG) : s;
        const bool active = (t >= 0) && (t < T);

        if (active) {
            // ---- layer-1: chunked WMMA GEMM for the input-gate contribution
            if (l1 && ((t & (LAG - 1)) == 0)) {
                if (wv < 4) {                 // wave-uniform; EXEC all ones
                    const int n0  = wv * 16;  // local gate-row tile
                    const int mA  = lane & 15, kbA = (lane >> 4) * 8;
                    const int nB  = lane & 15, kbB = (lane >> 4) * 8;
                    const int nLoc = n0 + nB;
                    const int Rg  = (nLoc >> 4) * HD + wid * UPW + (nLoc & 15);
                    float bias = bih1[Rg] + bhh1[Rg];
                    v8f c;
#pragma unroll
                    for (int i = 0; i < 8; ++i) c[i] = bias;
                    const _Float16* arow = h1h + (size_t)(t + mA) * HD;
                    const _Float16* brow = wih1h + (size_t)Rg * HD;
                    for (int k0 = 0; k0 < HD; k0 += 32) {
                        v8h a0 = *(const v8h*)(arow + k0 + kbA);
                        v8h a1 = *(const v8h*)(arow + k0 + kbA + 16);
                        v8h b0 = *(const v8h*)(brow + k0 + kbB);
                        v8h b1 = *(const v8h*)(brow + k0 + kbB + 16);
                        v16h av = __builtin_shufflevector(a0, a1, 0,1,2,3,4,5,6,7,
                                                          8,9,10,11,12,13,14,15);
                        v16h bv = __builtin_shufflevector(b0, b1, 0,1,2,3,4,5,6,7,
                                                          8,9,10,11,12,13,14,15);
                        c = __builtin_amdgcn_wmma_f32_16x16x32_f16(
                                false, av, false, bv, (short)0, c, false, false);
                    }
                    const int mD = (lane >> 4) * 8, nD = lane & 15;
#pragma unroll
                    for (int rr = 0; rr < 8; ++rr)
                        gxb[(mD + rr) * ROWS + n0 + nD] = c[rr];
                }
                __syncthreads();              // gxb visible to whole WG
            }

            // ---- stage current h into LDS (t==0 reads the zeroed buffer)
            const float* hin = myh + (t & 1) * HD;
            for (int i = tid; i < HD; i += 256) hl[i] = hin[i];
            __syncthreads();

            // ---- g[r] = W_slice[r]·h   (4 lanes x 256 strided MACs)
            const float* wr = wl + r * WSTR;
            float acc = 0.0f;
#pragma unroll 8
            for (int kk = 0; kk < 256; ++kk) {
                int k = sub + (kk << 2);
                acc = fmaf(wr[k], hl[k], acc);
            }
            acc += __shfl_xor(acc, 1, 4);
            acc += __shfl_xor(acc, 2, 4);
            if (sub == 0) {
                float g0 = l1 ? gxb[(t & (LAG - 1)) * ROWS + r]
                              : gx0[(size_t)t * H4 + Rglob];
                gl[r] = acc + g0;
            }
            __syncthreads();

            // ---- cell update for this WG's UPW units (gate order i,f,g,o)
            if (tid < UPW) {
                float i_ = sigmoidf_(gl[tid]);
                float f_ = sigmoidf_(gl[UPW + tid]);
                float g_ = tanhf(gl[2 * UPW + tid]);
                float o_ = sigmoidf_(gl[3 * UPW + tid]);
                c_reg = fmaf(f_, c_reg, i_ * g_);
                float h_ = o_ * tanhf(c_reg);
                int u = wid * UPW + tid;
                myh[((t + 1) & 1) * HD + u] = h_;
                if (!l1) h1h[(size_t)t * HD + u] = (_Float16)h_;
            }
            __syncthreads();
        }

        // ---- device-scope barrier: monotone epoch counter, no reset race
        if (tid == 0) {
            __threadfence();
            atomicAdd(counter, 1u);
            unsigned target = (unsigned)(s + 1) * NWGT;
            while (__hip_atomic_load(counter, __ATOMIC_RELAXED,
                                     __HIP_MEMORY_SCOPE_AGENT) < target)
                __builtin_amdgcn_s_sleep(2);
        }
        __syncthreads();
        __threadfence();                      // acquire: see other WGs' writes
    }
}

// ---------------------------------------------------------------------------
// out = h_last · w_lin + b_lin        (OUT == 1)
// ---------------------------------------------------------------------------
__global__ void final_linear(const float* __restrict__ h_last,
                             const float* __restrict__ w_lin,
                             const float* __restrict__ b_lin,
                             float* __restrict__ out) {
    __shared__ float red[256];
    int tid = threadIdx.x;
    float s = 0.0f;
    for (int k = tid; k < HD; k += 256) s = fmaf(h_last[k], w_lin[k], s);
    red[tid] = s;
    __syncthreads();
    for (int off = 128; off > 0; off >>= 1) {
        if (tid < off) red[tid] += red[tid + off];
        __syncthreads();
    }
    if (tid == 0) out[0] = red[0] + b_lin[0];
}

// ---------------------------------------------------------------------------
extern "C" void kernel_launch(void* const* d_in, const int* in_sizes, int n_in,
                              void* d_out, int out_size, void* d_ws, size_t ws_size,
                              hipStream_t stream) {
    const float* x    = (const float*)d_in[0];
    const float* wih0 = (const float*)d_in[1];
    const float* whh0 = (const float*)d_in[2];
    const float* bih0 = (const float*)d_in[3];
    const float* bhh0 = (const float*)d_in[4];
    const float* wih1 = (const float*)d_in[5];
    const float* whh1 = (const float*)d_in[6];
    const float* bih1 = (const float*)d_in[7];
    const float* bhh1 = (const float*)d_in[8];
    const float* wlin = (const float*)d_in[9];
    const float* blin = (const float*)d_in[10];

    const int T = in_sizes[0] / 9;            // 8192

    // Workspace carve-up (all 256B aligned).
    char* p = (char*)d_ws;
    auto take = [&](size_t bytes) {
        char* q = p;
        p += (bytes + 255) & ~(size_t)255;
        return (void*)q;
    };
    float*     gx0   = (float*)take((size_t)T * H4 * 4);
    _Float16*  h1h   = (_Float16*)take((size_t)T * HD * 2);
    _Float16*  wih1h = (_Float16*)take((size_t)H4 * HD * 2);
    float*     hbuf0 = (float*)take(2 * HD * 4);
    float*     hbuf2 = (float*)take(2 * HD * 4);
    unsigned*  cnt   = (unsigned*)take(sizeof(unsigned));

    const size_t fusedLds =
        (size_t)(ROWS * WSTR + HD + ROWS + LAG * ROWS) * sizeof(float);

    init_state<<<1, 256, 0, stream>>>(hbuf0, hbuf2, cnt);

    {   // layer-0 input precompute (K=9)
        size_t n = (size_t)T * H4;
        gx0_kernel<<<(unsigned)((n + 255) / 256), 256, 0, stream>>>(
            x, wih0, bih0, bhh0, gx0, T);
    }
    {   // w_ih1 -> f16 for the in-kernel WMMA chunks
        size_t n = (size_t)H4 * HD;
        cvt_f16<<<(unsigned)((n + 255) / 256), 256, 0, stream>>>(wih1, wih1h, n);
    }

    // Fused 2-layer pipelined scan: T+LAG grid barriers total.
    lstm_fused<<<NWGT, 256, fusedLds, stream>>>(
        whh0, gx0, whh1, wih1h, bih1, bhh1, h1h, hbuf0, hbuf2, cnt, T);

    // Last layer-1 hidden state lives at parity (T & 1) of its ping-pong.
    const float* hlast = hbuf2 + (size_t)(T & 1) * HD;
    final_linear<<<1, 256, 0, stream>>>(hlast, wlin, blin, (float*)d_out);
}
